// baseLSNN_20280835571712
// MI455X (gfx1250) — matrix operand order repaired and checked by reading, not compiled, not run on hardware
//
#include <hip/hip_runtime.h>

// LSNN forward scan for MI455X (gfx1250), wave32 + WMMA bf16.
// B=128, T=1000, n_in=256, n_rec=512.
// Grid: 8 workgroups (one per 16-row batch tile) x 1024 threads (32 waves,
// one 16-column neuron tile per wave). Per-step sync = one workgroup barrier.
// Weights are bf16 in d_ws, re-read from L2 every step. The (uniform) base
// pointers are laundered through an SGPR asm each iteration so the compiler
// cannot hoist 384 VGPRs of loop-invariant B operands and spill them to
// scratch (round-1 bug); the divergent per-lane offset is added AFTER the
// launder (round-3's "+s" on a lane-varying value caused an illegal
// VGPR->SGPR copy). The address_space(1) cast keeps loads on the GLOBAL path
// (LOADcnt only) instead of FLAT (LOADcnt+DScnt) as round-2 emitted.

typedef __attribute__((ext_vector_type(16))) __bf16 v16bf;
typedef __attribute__((ext_vector_type(8)))  __bf16 v8bf;
typedef __attribute__((ext_vector_type(8)))  float  v8f;

typedef const __bf16 __attribute__((address_space(1)))* gbf16p;
typedef const v8bf  __attribute__((address_space(1)))* gv8bfp;

#define BSZ   128
#define TSTEP 1000
#define NIN   256
#define NREC  512

#define DECAY_V 0.951229424500714f   /* exp(-1/20) */
#define DECAY_A 0.951229424500714f
#define ONE_M_DA 0.048770575499286f
#define THR_C   0.01f
#define BETA_C  1.8f

// ---- prologue: fp32 -> bf16 weight conversion ------------------------------

__global__ void cvt_win_kernel(const float* __restrict__ w, __bf16* __restrict__ o) {
    int i = blockIdx.x * blockDim.x + threadIdx.x;
    if (i < NIN * NREC) o[i] = (__bf16)w[i];
}

__global__ void cvt_wrec_kernel(const float* __restrict__ w, __bf16* __restrict__ o) {
    int i = blockIdx.x * blockDim.x + threadIdx.x;
    if (i < NREC * NREC) {
        int r = i / NREC, c = i % NREC;
        float v = (r == c) ? 0.0f : w[i];   // zero diagonal (1 - eye mask)
        o[i] = (__bf16)v;
    }
}

// ---- main persistent scan kernel -------------------------------------------

__launch_bounds__(1024, 1)
__global__ void lsnn_scan_kernel(const float*  __restrict__ x,
                                 const __bf16* __restrict__ winb,
                                 const __bf16* __restrict__ wrb,
                                 float*        __restrict__ out)
{
    __shared__ __bf16 x_lds[2][16 * NIN];    // 2 x 8 KB, staged x(t) as bf16
    __shared__ __bf16 z_lds[2][16 * NREC];   // 2 x 16 KB, z_t double buffer

    const int tid   = threadIdx.x;
    const int lane  = tid & 31;
    const int wave  = tid >> 5;              // 0..31 -> neuron tile
    const int btile = blockIdx.x;            // 0..7  -> batch tile
    const int ncol0 = wave * 16;

    // A-matrix (16x32 bf16) per-lane addressing (ISA 7.12.2):
    //   lanes 0-15:  row M=lane,    K-chunks [0..7] and [16..23]
    //   lanes 16-31: row M=lane-16, K-chunks [8..15] and [24..31]
    const int mA    = lane & 15;
    const int hiA   = lane >> 4;
    const int kofsA = hiA * 8;

    // per-lane weight offset (bf16 elements): row K=8*hiA, col N=ncol0+mA
    // (divergent -> must stay out of the SGPR-laundered value)
    const size_t wofs = (size_t)kofsA * NREC + ncol0 + mA;

    // init z_{-1} = 0 in buffer 1 (t=0 reads buffer (0-1)&1 = 1)
    for (int i = tid; i < 16 * NREC; i += 1024) z_lds[1][i] = (__bf16)0.0f;

    // per-lane register state: 8 C/D elements (M = r + 8*hiA, N = ncol0 + mA)
    float v[8], a[8], zp[8];
    #pragma unroll
    for (int r = 0; r < 8; ++r) { v[r] = 0.0f; a[r] = 0.0f; zp[r] = 0.0f; }

    // cooperative x-staging: 16 rows x 256 cols fp32, 4 floats per thread
    const int xrow = tid >> 6;               // 0..15
    const int xcol = (tid & 63) << 2;        // 0..252 step 4
    const float* xbase = x + ((size_t)(btile * 16 + xrow) * TSTEP) * NIN + xcol;

    for (int t = 0; t < TSTEP; ++t) {
        const int buf = t & 1;

        // stage x(t) -> bf16 LDS (A-row-contiguous layout)
        {
            const float4 xv = *(const float4*)(xbase + (size_t)t * NIN);
            __bf16* dst = &x_lds[buf][xrow * NIN + xcol];
            dst[0] = (__bf16)xv.x; dst[1] = (__bf16)xv.y;
            dst[2] = (__bf16)xv.z; dst[3] = (__bf16)xv.w;
        }
        if (t + 1 < TSTEP)
            __builtin_prefetch(xbase + (size_t)(t + 1) * NIN, 0, 1);

        // one barrier per step:
        //  - publishes x(t) and z_{t-1} (written at end of prev iter)
        //  - separates prev iter's reads of z_lds[buf] from this iter's writes
        __syncthreads();

        // Launder the UNIFORM weight base pointers in SGPRs (opaque per
        // iteration -> no cross-loop hoist/spill); add the divergent per-lane
        // offset afterwards on the address_space(1) pointer.
        unsigned long long winp = (unsigned long long)winb;
        unsigned long long wrp  = (unsigned long long)wrb;
        asm volatile("" : "+s"(winp), "+s"(wrp));
        gbf16p winb_t = (gbf16p)winp + wofs;
        gbf16p wrb_t  = (gbf16p)wrp  + wofs;

        v8f acc0 = {};  // two independent accumulator chains
        v8f acc1 = {};

        // fused input GEMM: i_t = x(t) @ w_in, K = 256 (8 WMMAs)
        #pragma unroll
        for (int kc = 0; kc < NIN / 32; ++kc) {
            const int k0 = kc * 32 + kofsA;
            v8bf alo = *(const v8bf*)&x_lds[buf][mA * NIN + k0];
            v8bf ahi = *(const v8bf*)&x_lds[buf][mA * NIN + k0 + 16];
            v16bf A = __builtin_shufflevector(alo, ahi,
                0,1,2,3,4,5,6,7,8,9,10,11,12,13,14,15);
            gbf16p bp = winb_t + (size_t)kc * 32 * NREC;
            v16bf Bm;
            #pragma unroll
            for (int s = 0; s < 16; ++s)
                Bm[s] = bp[(size_t)((s & 7) + (s >> 3) * 16) * NREC];
            if (kc & 1)
                acc1 = __builtin_amdgcn_wmma_f32_16x16x32_bf16(
                    false, A, false, Bm, (short)0, acc1, false, false);
            else
                acc0 = __builtin_amdgcn_wmma_f32_16x16x32_bf16(
                    false, A, false, Bm, (short)0, acc0, false, false);
        }

        // recurrent GEMM: z_{t-1} @ wr, K = 512 (16 WMMAs)
        const __bf16* zsrc = &z_lds[buf ^ 1][0];
        #pragma unroll
        for (int kc = 0; kc < NREC / 32; ++kc) {
            const int k0 = kc * 32 + kofsA;
            v8bf alo = *(const v8bf*)&zsrc[mA * NREC + k0];
            v8bf ahi = *(const v8bf*)&zsrc[mA * NREC + k0 + 16];
            v16bf A = __builtin_shufflevector(alo, ahi,
                0,1,2,3,4,5,6,7,8,9,10,11,12,13,14,15);
            gbf16p bp = wrb_t + (size_t)kc * 32 * NREC;
            v16bf Bm;
            #pragma unroll
            for (int s = 0; s < 16; ++s)
                Bm[s] = bp[(size_t)((s & 7) + (s >> 3) * 16) * NREC];
            if (kc & 1)
                acc1 = __builtin_amdgcn_wmma_f32_16x16x32_bf16(
                    false, A, false, Bm, (short)0, acc1, false, false);
            else
                acc0 = __builtin_amdgcn_wmma_f32_16x16x32_bf16(
                    false, A, false, Bm, (short)0, acc0, false, false);
        }

        // elementwise LIF-with-adaptation update on 8 C/D elements per lane
        __bf16* zdst = &z_lds[buf][0];
        #pragma unroll
        for (int r = 0; r < 8; ++r) {
            const int M = r + 8 * hiA;                 // batch row in tile
            const float iin = acc0[r] + acc1[r];
            float na   = DECAY_A * a[r] + ONE_M_DA * zp[r];
            float athr = THR_C + BETA_C * na;
            float nv   = DECAY_V * v[r] + iin - athr * zp[r];  // soft reset
            float vsc  = nv - athr;
            float nz   = (vsc > 0.0f) ? 1.0f : 0.0f;
            v[r] = nv; a[r] = na; zp[r] = nz;
            zdst[M * NREC + ncol0 + mA] = (__bf16)nz;  // publish z_t
            // spikes stream 262 MB write-once: non-temporal so they do not
            // evict the L2-hot weights / x stream
            __builtin_nontemporal_store(nz,
                out + ((size_t)(btile * 16 + M) * TSTEP + t) * NREC + ncol0 + mA);
        }
        // z_t writes are ordered before next iteration's __syncthreads()
    }
}

// ---- launch ----------------------------------------------------------------

extern "C" void kernel_launch(void* const* d_in, const int* in_sizes, int n_in,
                              void* d_out, int out_size, void* d_ws, size_t ws_size,
                              hipStream_t stream) {
    const float* x     = (const float*)d_in[0];
    const float* w_in  = (const float*)d_in[1];
    const float* w_rec = (const float*)d_in[2];
    float* out = (float*)d_out;

    // workspace: bf16 weights (w_in: 256 KB, w_rec: 512 KB)
    __bf16* winb = (__bf16*)d_ws;
    __bf16* wrb  = (__bf16*)((char*)d_ws + (size_t)NIN * NREC * sizeof(__bf16));

    hipLaunchKernelGGL(cvt_win_kernel,  dim3((NIN * NREC + 255) / 256),  dim3(256), 0, stream, w_in,  winb);
    hipLaunchKernelGGL(cvt_wrec_kernel, dim3((NREC * NREC + 255) / 256), dim3(256), 0, stream, w_rec, wrb);
    hipLaunchKernelGGL(lsnn_scan_kernel, dim3(BSZ / 16), dim3(1024), 0, stream, x, winb, wrb, out);
}
